// MoELayer_19327352832722
// MI455X (gfx1250) — matrix operand order repaired and checked by reading, not compiled, not run on hardware
//
#include <hip/hip_runtime.h>
#include <hip/hip_fp16.h>
#include <math.h>

#define HID  768
#define FFN  3072
#define NE   8
#define NTOK 1024
#define TOPK 2
#define PADT 64            // pad each expert's token count to a multiple of 64 (4 WMMA tiles)
#define CAPS 2560          // 2048 pairs + up to 8*63 padding, rounded up

typedef _Float16 half_t;
typedef __attribute__((ext_vector_type(16))) _Float16 v16h;
typedef __attribute__((ext_vector_type(8)))  _Float16 v8h;
typedef __attribute__((ext_vector_type(4)))  _Float16 v4h;
typedef __attribute__((ext_vector_type(8)))  float    v8f;
typedef __attribute__((ext_vector_type(4)))  float    v4f;

struct WS {
    int   counts[NE];
    int   cursors[NE];
    int   bases[NE];
    int   padded[NE];
    int   top_e[NTOK * TOPK];
    float top_w[NTOK * TOPK];
    int   slot_of[NTOK * TOPK];
    int   tok_of_slot[CAPS];
};

// branch-free GELU (tanh form): 0.5x(1+tanh(0.79788456(x+0.044715x^3)))
// tanh(t) = 1 - 2*rcp(exp(2t)+1); saturates correctly via exp->inf / ->0.
__device__ __forceinline__ float gelu_f(float v) {
    float t  = 0.7978845608f * (v + 0.044715f * v * v * v);
    float u  = __expf(2.0f * t);
    float th = 1.0f - 2.0f * __builtin_amdgcn_rcpf(u + 1.0f);
    return 0.5f * v * (1.0f + th);
}

// ---------------- init ----------------
__global__ void k_init(WS* ws) {
    int t = threadIdx.x;
    if (t < NE) { ws->counts[t] = 0; ws->cursors[t] = 0; }
}

// ---------------- router: logits -> softmax -> top2 -> renorm ----------------
__global__ void k_router(const float* __restrict__ x, const float* __restrict__ rw, WS* ws) {
    int n = blockIdx.x * blockDim.x + threadIdx.x;
    if (n >= NTOK) return;
    const float* xr = x + (size_t)n * HID;
    float lg[NE];
#pragma unroll
    for (int e = 0; e < NE; ++e) lg[e] = 0.0f;
    for (int h = 0; h < HID; ++h) {
        float xv = xr[h];
        const float* r = rw + (size_t)h * NE;
#pragma unroll
        for (int e = 0; e < NE; ++e) lg[e] += xv * r[e];
    }
    float m = lg[0];
#pragma unroll
    for (int e = 1; e < NE; ++e) m = fmaxf(m, lg[e]);
    float p[NE];
#pragma unroll
    for (int e = 0; e < NE; ++e) p[e] = __expf(lg[e] - m); // softmax denom cancels in top-k renorm
    int i0 = 0;
#pragma unroll
    for (int e = 1; e < NE; ++e) if (p[e] > p[i0]) i0 = e; // strict > keeps lowest index, like top_k
    int i1 = (i0 == 0) ? 1 : 0;
#pragma unroll
    for (int e = 0; e < NE; ++e) if (e != i0 && p[e] > p[i1]) i1 = e;
    float w0 = p[i0], w1 = p[i1], t = w0 + w1;
    w0 /= t; w1 /= t;
    ws->top_e[2 * n + 0] = i0;  ws->top_e[2 * n + 1] = i1;
    ws->top_w[2 * n + 0] = w0;  ws->top_w[2 * n + 1] = w1;
    atomicAdd(&ws->counts[i0], 1);
    atomicAdd(&ws->counts[i1], 1);
}

// ---------------- finalize: padded counts, bases, pad slots ----------------
__global__ void k_finalize(WS* ws) {
    if (threadIdx.x != 0 || blockIdx.x != 0) return;
    int base = 0;
    for (int e = 0; e < NE; ++e) {
        int c  = ws->counts[e];
        int pc = (c + PADT - 1) / PADT * PADT;
        ws->bases[e]  = base;
        ws->padded[e] = pc;
        for (int s = c; s < pc; ++s) ws->tok_of_slot[base + s] = 0; // dummy token, weight unused
        base += pc;
    }
}

// ---------------- scatter tokens into per-expert slot lists ----------------
__global__ void k_scatter(WS* ws) {
    int n = blockIdx.x * blockDim.x + threadIdx.x;
    if (n >= NTOK) return;
#pragma unroll
    for (int k = 0; k < TOPK; ++k) {
        int e    = ws->top_e[2 * n + k];
        int pos  = atomicAdd(&ws->cursors[e], 1);
        int slot = ws->bases[e] + pos;
        ws->tok_of_slot[slot]   = n;
        ws->slot_of[2 * n + k]  = slot;
    }
}

// ---------------- x fp32 -> f16 (vectorized: 4 per thread) ----------------
__global__ void k_cvt(const float* __restrict__ x, half_t* __restrict__ xh) {
    int i = blockIdx.x * blockDim.x + threadIdx.x;
    v4f v = *(const v4f*)(x + 4 * (size_t)i);
    v4h h;
#pragma unroll
    for (int j = 0; j < 4; ++j) h[j] = (half_t)v[j];
    *(v4h*)(xh + 4 * (size_t)i) = h;
}

// ---------------- FFN1: h = gelu(Xg @ W1[e]), f16 out ----------------
__global__ void __launch_bounds__(256) k_ffn1(const float* __restrict__ w1,
                                              const half_t* __restrict__ xh,
                                              half_t* __restrict__ h_ws,
                                              const WS* __restrict__ ws) {
    const int bid  = blockIdx.x;
    const int e    = bid / (FFN / 128);
    const int ct   = bid % (FFN / 128);
    const int wave = threadIdx.x >> 5;
    const int lane = threadIdx.x & 31;
    const int hl   = lane >> 4;          // K-half of wave
    const int lr   = lane & 15;          // row/col index within fragment
    const int col  = ct * 128 + wave * 16 + lr;
    const float* w1e = w1 + (size_t)e * HID * FFN;
    const int base = ws->bases[e];
    const int ntok = ws->padded[e];

    for (int tt = 0; tt < ntok; tt += PADT) {
        int tok[4];
#pragma unroll
        for (int j = 0; j < 4; ++j) tok[j] = ws->tok_of_slot[base + tt + j * 16 + lr];
        v8f acc[4];
#pragma unroll
        for (int j = 0; j < 4; ++j) acc[j] = (v8f)(0.0f);

        for (int kc = 0; kc < HID; kc += 32) {
            v16h b;
#pragma unroll
            for (int i = 0; i < 16; ++i)
                b[i] = (half_t)w1e[(size_t)(kc + hl * 8 + (i & 7) + ((i >> 3) << 4)) * FFN + col];
#pragma unroll
            for (int j = 0; j < 4; ++j) {
                const half_t* p = xh + (size_t)tok[j] * HID + kc + hl * 8;
                v8h alo = *(const v8h*)(p);
                v8h ahi = *(const v8h*)(p + 16);
                v16h a;
#pragma unroll
                for (int i = 0; i < 8; ++i) { a[i] = alo[i]; a[i + 8] = ahi[i]; }
                acc[j] = __builtin_amdgcn_wmma_f32_16x16x32_f16(
                    false, a, false, b, (short)0, acc[j], false, false);
            }
        }
        // epilogue: branch-free GELU, store f16
#pragma unroll
        for (int j = 0; j < 4; ++j) {
#pragma unroll
            for (int r = 0; r < 8; ++r) {
                float v = gelu_f(acc[j][r]);
                int slot = base + tt + j * 16 + r + hl * 8;
                h_ws[(size_t)slot * FFN + col] = (half_t)v;
            }
        }
    }
}

// ---------------- FFN2: y = H @ W2[e], fp32 per-slot rows ----------------
__global__ void __launch_bounds__(256) k_ffn2(const float* __restrict__ w2,
                                              const half_t* __restrict__ h_ws,
                                              float* __restrict__ y_ws,
                                              const WS* __restrict__ ws) {
    const int bid  = blockIdx.x;
    const int e    = bid / (HID / 128);
    const int ct   = bid % (HID / 128);
    const int wave = threadIdx.x >> 5;
    const int lane = threadIdx.x & 31;
    const int hl   = lane >> 4;
    const int lr   = lane & 15;
    const int col  = ct * 128 + wave * 16 + lr;
    const float* w2e = w2 + (size_t)e * FFN * HID;
    const int base = ws->bases[e];
    const int ntok = ws->padded[e];

    for (int tt = blockIdx.y * PADT; tt < ntok; tt += 4 * PADT) {
        v8f acc[4];
#pragma unroll
        for (int j = 0; j < 4; ++j) acc[j] = (v8f)(0.0f);

        for (int kc = 0; kc < FFN; kc += 32) {
            v16h b;
#pragma unroll
            for (int i = 0; i < 16; ++i)
                b[i] = (half_t)w2e[(size_t)(kc + hl * 8 + (i & 7) + ((i >> 3) << 4)) * HID + col];
#pragma unroll
            for (int j = 0; j < 4; ++j) {
                const half_t* p = h_ws + (size_t)(base + tt + j * 16 + lr) * FFN + kc + hl * 8;
                v8h alo = *(const v8h*)(p);
                v8h ahi = *(const v8h*)(p + 16);
                v16h a;
#pragma unroll
                for (int i = 0; i < 8; ++i) { a[i] = alo[i]; a[i + 8] = ahi[i]; }
                acc[j] = __builtin_amdgcn_wmma_f32_16x16x32_f16(
                    false, a, false, b, (short)0, acc[j], false, false);
            }
        }
#pragma unroll
        for (int j = 0; j < 4; ++j) {
#pragma unroll
            for (int r = 0; r < 8; ++r) {
                int slot = base + tt + j * 16 + r + hl * 8;
                y_ws[(size_t)slot * HID + col] = acc[j][r];
            }
        }
    }
}

// ---------------- combine: out[n] = w0*y[slot0] + w1*y[slot1] (vec4) ----------------
__global__ void k_combine(const WS* __restrict__ ws, const float* __restrict__ y_ws,
                          float* __restrict__ out) {
    int n  = blockIdx.x;
    int s0 = ws->slot_of[2 * n + 0];
    int s1 = ws->slot_of[2 * n + 1];
    float w0 = ws->top_w[2 * n + 0];
    float w1 = ws->top_w[2 * n + 1];
    const v4f* y0 = (const v4f*)(y_ws + (size_t)s0 * HID);
    const v4f* y1 = (const v4f*)(y_ws + (size_t)s1 * HID);
    v4f* o = (v4f*)(out + (size_t)n * HID);
    int c = threadIdx.x;                 // 192 vec4 = 768 floats
    if (c < HID / 4) o[c] = w0 * y0[c] + w1 * y1[c];
}

extern "C" void kernel_launch(void* const* d_in, const int* in_sizes, int n_in,
                              void* d_out, int out_size, void* d_ws, size_t ws_size,
                              hipStream_t stream) {
    const float* x  = (const float*)d_in[0];
    const float* rw = (const float*)d_in[1];
    const float* w1 = (const float*)d_in[2];
    const float* w2 = (const float*)d_in[3];
    float* out = (float*)d_out;

    char* wsb = (char*)d_ws;
    WS* ws = (WS*)wsb;
    size_t off = (sizeof(WS) + 255) & ~(size_t)255;
    half_t* x_h = (half_t*)(wsb + off);
    off += (size_t)NTOK * HID * sizeof(half_t); off = (off + 255) & ~(size_t)255;
    half_t* h_ws = (half_t*)(wsb + off);
    off += (size_t)CAPS * FFN * sizeof(half_t); off = (off + 255) & ~(size_t)255;
    float* y_ws = (float*)(wsb + off);

    k_init    <<<1, 32, 0, stream>>>(ws);
    k_router  <<<(NTOK + 255) / 256, 256, 0, stream>>>(x, rw, ws);
    k_finalize<<<1, 1, 0, stream>>>(ws);
    k_scatter <<<(NTOK + 255) / 256, 256, 0, stream>>>(ws);
    k_cvt     <<<(NTOK * HID / 4) / 256, 256, 0, stream>>>(x, x_h);
    k_ffn1    <<<NE * (FFN / 128), 256, 0, stream>>>(w1, x_h, h_ws, ws);
    k_ffn2    <<<dim3(NE * (HID / 128), 4), 256, 0, stream>>>(w2, h_ws, y_ws, ws);
    k_combine <<<NTOK, 256, 0, stream>>>(ws, y_ws, out);
}
